// Discriminator_55757265437038
// MI455X (gfx1250) — matrix operand order repaired and checked by reading, not compiled, not run on hardware
//
#include <hip/hip_runtime.h>
#include <hip/hip_bf16.h>

// ---------------------------------------------------------------------------
// CDNA5 (gfx1250) GAN discriminator. Channels-last (NHWC) implicit GEMM on
// v_wmma_f32_16x16x32_f16 (wave32). K-order is (kh, kw, ci) so every im2col
// B-fragment is 16 *contiguous* f16 values (activations are pre-padded, so no
// bounds checks). Wave tile: 64(Cout) x 32(cols) = 4x2 fragments, 8 WMMA/step.
// ---------------------------------------------------------------------------

typedef __attribute__((ext_vector_type(16))) _Float16 v16h;
typedef __attribute__((ext_vector_type(8)))  float    v8f;

// 16 contiguous halves. C%8==0 -> 32B-aligned (2xb128); C==4 -> 8B (4xb64).
template <int C>
__device__ __forceinline__ v16h load_contig16(const _Float16* __restrict__ p) {
    if constexpr (C % 8 == 0) {
        union { uint4 u[2]; v16h h; } t;
        t.u[0] = *reinterpret_cast<const uint4*>(p);
        t.u[1] = *reinterpret_cast<const uint4*>(p + 8);
        return t.h;
    } else {
        union { uint2 u[4]; v16h h; } t;
#pragma unroll
        for (int j = 0; j < 4; ++j)
            t.u[j] = *reinterpret_cast<const uint2*>(p + j * 4);
        return t.h;
    }
}

// A-fragment (weights, row-major [Cout][K]); per-lane K-offsets ka..ka+15
// (contiguous, matching B) -> two aligned b128 loads.
__device__ __forceinline__ v16h load_afrag(const _Float16* __restrict__ p) {
    union { uint4 u[2]; v16h h; } t;
    t.u[0] = *reinterpret_cast<const uint4*>(p);
    t.u[1] = *reinterpret_cast<const uint4*>(p + 8);
    return t.h;
}

// Implicit-GEMM conv, channels-last.
//   x : padded NHWC f16 [N][Hp][Wp][C] (zero borders already materialized)
//   w : f16 [Cout][KH][KW][C]
//   y32: raw NHWC f32 [N][Ho][Wo][Cout]            (if y16 == nullptr)
//   y16: padded NHWC f16 [N][HpO][WpO][Cout], fused LeakyReLU (if != nullptr)
// Requires: Cout%64==0, (N*Ho*Wo)%32==0, K%32==0, 16 | KW*C.
template <int C, int KH, int KW>
__global__ __launch_bounds__(32)
void conv_wmma_cl(const _Float16* __restrict__ x, const _Float16* __restrict__ w,
                  float* __restrict__ y32, _Float16* __restrict__ y16,
                  int N, int Hp, int Wp, int Cout, int Ho, int Wo,
                  int sh, int sw, int HpO, int WpO, int pt, int pl)
{
    constexpr int K   = C * KH * KW;
    constexpr int ROW = KW * C;            // K-elements per kh row

    const int lane  = threadIdx.x;
    const int col16 = lane & 15;
    const int half  = lane >> 4;

    const int coutBase = blockIdx.x * 64;
    const int colBase  = blockIdx.y * 32;

    // This lane's two output columns (n, ho, wo) and patch base pointers.
    int n_[2], ho_[2], wo_[2];
    const _Float16* base_[2];
#pragma unroll
    for (int ic = 0; ic < 2; ++ic) {
        int s  = colBase + ic * 16 + col16;
        int wo = s % Wo;
        int t  = s / Wo;
        int ho = t % Ho;
        int nn = t / Ho;
        n_[ic] = nn; ho_[ic] = ho; wo_[ic] = wo;
        base_[ic] = x + (((size_t)nn * Hp + ho * sh) * Wp + wo * sw) * C;
    }

    const _Float16* wrow[4];
#pragma unroll
    for (int ia = 0; ia < 4; ++ia)
        wrow[ia] = w + (size_t)(coutBase + ia * 16 + col16) * K;

    v8f acc[4][2];
#pragma unroll
    for (int ia = 0; ia < 4; ++ia)
#pragma unroll
        for (int ic = 0; ic < 2; ++ic)
            acc[ia][ic] = (v8f){};

#pragma unroll 2
    for (int k0 = 0; k0 < K; k0 += 32) {
        const int ka = k0 + half * 16;   // A-fragment K offset
        const int kb = k0 + half * 16;   // B-fragment K offset

        // Decompose kb once per fragment (compile-time divisors).
        const int kh  = kb / ROW;
        const int r   = kb - kh * ROW;
        const int kw0 = r / C;
        const int ci0 = r - kw0 * C;
        const size_t boff = ((size_t)kh * Wp + kw0) * C + ci0;

        v16h a[4];
#pragma unroll
        for (int ia = 0; ia < 4; ++ia)
            a[ia] = load_afrag(wrow[ia] + ka);

        v16h b[2];
#pragma unroll
        for (int ic = 0; ic < 2; ++ic)
            b[ic] = load_contig16<C>(base_[ic] + boff);

#pragma unroll
        for (int ia = 0; ia < 4; ++ia)
#pragma unroll
            for (int ic = 0; ic < 2; ++ic)
                acc[ia][ic] = __builtin_amdgcn_wmma_f32_16x16x32_f16(
                    false, a[ia], false, b[ic], (short)0, acc[ia][ic], false, false);
    }

    // Epilogue: VGPR i holds M = half*8 + i, column = lane & 15.
    // NHWC: 8 consecutive channels per lane -> single vector store.
#pragma unroll
    for (int ia = 0; ia < 4; ++ia) {
#pragma unroll
        for (int ic = 0; ic < 2; ++ic) {
            const int nn = n_[ic], ho = ho_[ic], wo = wo_[ic];
            const int m0 = coutBase + ia * 16 + half * 8;
            if (y16) {
                union { uint4 u; _Float16 h[8]; } t;
#pragma unroll
                for (int i = 0; i < 8; ++i) {
                    float v = acc[ia][ic][i];
                    v = (v >= 0.f) ? v : 0.2f * v;   // fused LeakyReLU
                    t.h[i] = (_Float16)v;
                }
                size_t oi = (((size_t)nn * HpO + ho + pt) * WpO + (wo + pl)) * Cout + m0;
                *reinterpret_cast<uint4*>(y16 + oi) = t.u;
            } else {
                size_t oi = (((size_t)nn * Ho + ho) * Wo + wo) * Cout + m0;
                float4 v0, v1;
                v0.x = acc[ia][ic][0]; v0.y = acc[ia][ic][1];
                v0.z = acc[ia][ic][2]; v0.w = acc[ia][ic][3];
                v1.x = acc[ia][ic][4]; v1.y = acc[ia][ic][5];
                v1.z = acc[ia][ic][6]; v1.w = acc[ia][ic][7];
                *reinterpret_cast<float4*>(y32 + oi)     = v0;
                *reinterpret_cast<float4*>(y32 + oi + 4) = v1;
            }
        }
    }
}

// ---------------------------------------------------------------------------
// Layout / conversion helpers
// ---------------------------------------------------------------------------

__global__ void fill_zero_f16(_Float16* __restrict__ p, size_t n) {
    size_t i = (size_t)blockIdx.x * blockDim.x + threadIdx.x;
    if (i < n) p[i] = (_Float16)0.f;
}

__global__ void cvt_f32_f16(const float* __restrict__ src, _Float16* __restrict__ dst, size_t n) {
    size_t i = (size_t)blockIdx.x * blockDim.x + threadIdx.x;
    if (i < n) dst[i] = (_Float16)src[i];
}

// Input image: NCHW f32 [64][3][376][278] -> NHWC4 f16 [64][376][278][4] (ch 3 = 0)
__global__ void reorder_input_kernel(const float* __restrict__ src, _Float16* __restrict__ dst,
                                     size_t total) {
    size_t i = (size_t)blockIdx.x * blockDim.x + threadIdx.x;
    if (i >= total) return;
    int c = (int)(i & 3);
    size_t r = i >> 2;
    int w = (int)(r % 278);
    size_t r2 = r / 278;
    int h = (int)(r2 % 376);
    int n = (int)(r2 / 376);
    float v = (c < 3) ? src[(((size_t)n * 3 + c) * 376 + h) * 278 + w] : 0.f;
    dst[i] = (_Float16)v;
}

// w1: OIHW f32 [64][3][8][8] -> [64][8][8][4] f16 (ci 3 = 0)
__global__ void reorder_w1_kernel(const float* __restrict__ src, _Float16* __restrict__ dst) {
    int i = blockIdx.x * blockDim.x + threadIdx.x;
    if (i >= 64 * 8 * 8 * 4) return;
    int ci = i & 3;
    int kw = (i >> 2) & 7;
    int kh = (i >> 5) & 7;
    int co = i >> 8;
    float v = (ci < 3) ? src[((co * 3 + ci) * 8 + kh) * 8 + kw] : 0.f;
    dst[i] = (_Float16)v;
}

// Generic OIHW f32 -> [Cout][KH][KW][C] f16
__global__ void reorder_w_kernel(const float* __restrict__ src, _Float16* __restrict__ dst,
                                 int C, int KH, int KW, size_t total) {
    size_t i = (size_t)blockIdx.x * blockDim.x + threadIdx.x;
    if (i >= total) return;
    int ci = (int)(i % C);
    size_t r = i / C;
    int kw = (int)(r % KW);
    size_t r2 = r / KW;
    int kh = (int)(r2 % KH);
    int co = (int)(r2 / KH);
    dst[i] = (_Float16)src[(((size_t)co * C + ci) * KH + kh) * KW + kw];
}

// Per-channel batch stats over NHWC f32 [NHW][C]; coalesced: 64 ch x 4 j-lanes.
__global__ __launch_bounds__(256)
void bn_stats_cl(const float* __restrict__ y, float* __restrict__ mean,
                 float* __restrict__ ivar, int NHW, int C)
{
    const int tc = threadIdx.x & 63;
    const int tj = threadIdx.x >> 6;
    const int c  = blockIdx.x * 64 + tc;
    float s = 0.f, q = 0.f;
    for (int j = tj; j < NHW; j += 4) {
        float v = y[(size_t)j * C + c];
        s += v; q += v * v;
    }
    __shared__ float ss[256], sq[256];
    ss[threadIdx.x] = s; sq[threadIdx.x] = q;
    __syncthreads();
    if (tj == 0) {
#pragma unroll
        for (int r = 1; r < 4; ++r) { s += ss[r * 64 + tc]; q += sq[r * 64 + tc]; }
        float inv = 1.f / (float)NHW;
        float m = s * inv;
        float v = q * inv - m * m;
        mean[c] = m;
        ivar[c] = rsqrtf(v + 1e-5f);
    }
}

// BN apply + LeakyReLU + f16, raw NHWC -> padded NHWC for the next conv.
__global__ void bn_lrelu_cl(const float* __restrict__ y,
                            const float* __restrict__ mean, const float* __restrict__ ivar,
                            const float* __restrict__ g, const float* __restrict__ b,
                            _Float16* __restrict__ out,
                            int C, int Ho, int Wo, int Hp, int Wp, int pt, int pl,
                            size_t total)
{
    size_t i = (size_t)blockIdx.x * blockDim.x + threadIdx.x;
    if (i >= total) return;
    int c = (int)(i % C);
    size_t r = i / C;
    int wo = (int)(r % Wo);
    size_t r2 = r / Wo;
    int ho = (int)(r2 % Ho);
    int n  = (int)(r2 / Ho);
    float v = (y[i] - mean[c]) * ivar[c] * g[c] + b[c];
    v = (v >= 0.f) ? v : 0.2f * v;
    out[(((size_t)n * Hp + ho + pt) * Wp + (wo + pl)) * C + c] = (_Float16)v;
}

// Msum[b1,i] = sum_b2 exp(-sum_j |M[b1,i,j] - M[b2,i,j]|); M is [64][4096]
__global__ __launch_bounds__(64)
void msum_kernel(const float* __restrict__ Mg, float* __restrict__ Msum)
{
    __shared__ float sM[64];
    __shared__ float red[64];
    const int i  = blockIdx.x;
    const int b1 = blockIdx.y;
    const int t  = threadIdx.x;   // b2

    sM[t] = Mg[(size_t)b1 * 4096 + i * 64 + t];
    __syncthreads();

    const float* row = Mg + (size_t)t * 4096 + i * 64;
    float d = 0.f;
#pragma unroll
    for (int j = 0; j < 64; ++j) d += fabsf(sM[j] - row[j]);
    red[t] = __expf(-d);
    __syncthreads();
    for (int s = 32; s > 0; s >>= 1) {
        if (t < s) red[t] += red[t + s];
        __syncthreads();
    }
    if (t == 0) Msum[b1 * 64 + i] = red[0];
}

// fc1: [64,256] x [128,256]^T -> [64,128]
__global__ __launch_bounds__(128)
void fc1_kernel(const float* __restrict__ feat, const float* __restrict__ w,
                float* __restrict__ out)
{
    const int b = blockIdx.x, o = threadIdx.x;
    const float* f  = feat + (size_t)b * 256;
    const float* wr = w + (size_t)o * 256;
    float acc = 0.f;
    for (int k = 0; k < 256; ++k) acc += f[k] * wr[k];
    out[b * 128 + o] = acc;
}

// label[b] = sigmoid([Msum | fc1out] . fc2_w)
__global__ __launch_bounds__(64)
void head_kernel(const float* __restrict__ msum, const float* __restrict__ fc1o,
                 const float* __restrict__ fc2w, float* __restrict__ label)
{
    const int b = threadIdx.x;
    float acc = 0.f;
    for (int i = 0; i < 64; ++i)  acc += msum[b * 64 + i] * fc2w[i];
    for (int o = 0; o < 128; ++o) acc += fc1o[b * 128 + o] * fc2w[64 + o];
    label[b] = 1.f / (1.f + __expf(-acc));
}

// ---------------------------------------------------------------------------
// Host launcher
// ---------------------------------------------------------------------------

extern "C" void kernel_launch(void* const* d_in, const int* in_sizes, int n_in,
                              void* d_out, int out_size, void* d_ws, size_t ws_size,
                              hipStream_t stream)
{
    (void)in_sizes; (void)n_in; (void)out_size; (void)ws_size;

    const float* x_in  = (const float*)d_in[0];
    const float* w1    = (const float*)d_in[1];
    const float* w2    = (const float*)d_in[2];
    const float* w3    = (const float*)d_in[3];
    const float* w4    = (const float*)d_in[4];
    const float* w5    = (const float*)d_in[5];
    const float* g2    = (const float*)d_in[6];
    const float* b2    = (const float*)d_in[7];
    const float* g3    = (const float*)d_in[8];
    const float* b3    = (const float*)d_in[9];
    const float* g4    = (const float*)d_in[10];
    const float* b4    = (const float*)d_in[11];
    const float* T     = (const float*)d_in[12];
    const float* fc1_w = (const float*)d_in[13];
    const float* fc2_w = (const float*)d_in[14];
    float* out = (float*)d_out;   // [0:16384) features, [16384:16448) labels

    const int Nb = 64;
    // conv1: 4(3)x376x278 -> 64x93x91, 8x8 s(4,3) p0     K=256
    // conv2: 64x95x95(p1,2)-> 128x31x31, 5x5 s(3,3)      K=1600
    // conv3: 128x35x35(p2) -> 256x11x11, 5x5 s(3,3)      K=3200
    // conv4: 256x13x13(p1) -> 512x4x4,  7x7 s(2,2)       K=12544
    // conv5: GEMM K=8192;  M-gemm: K=256, Cout=4096
    const size_t nX0p = (size_t)Nb * 376 * 278 * 4;        // 26,759,168
    const size_t nX1p = (size_t)Nb * 95 * 95 * 64;         // 36,966,400
    const size_t nY2  = (size_t)Nb * 31 * 31 * 128;        //  7,872,512
    const size_t nX2p = (size_t)Nb * 35 * 35 * 128;        // 10,035,200
    const size_t nY3  = (size_t)Nb * 11 * 11 * 256;        //  1,982,464
    const size_t nX3p = (size_t)Nb * 13 * 13 * 256;        //  2,768,896
    const size_t nY4  = (size_t)Nb * 4 * 4 * 512;          //    524,288
    const size_t nX4  = nY4;
    const size_t nW1r = (size_t)64 * 8 * 8 * 4;            //     16,384
    const size_t nW2  = (size_t)128 * 64 * 5 * 5;          //    204,800
    const size_t nW3  = (size_t)256 * 128 * 5 * 5;         //    819,200
    const size_t nW4  = (size_t)512 * 256 * 7 * 7;         //  6,422,528
    const size_t nW5  = (size_t)256 * 512 * 4 * 4;         //  2,097,152
    const size_t nT   = (size_t)4096 * 256;                //  1,048,576
    const size_t nFt  = (size_t)Nb * 256;                  //     16,384
    const size_t nM   = (size_t)Nb * 4096;                 //    262,144

    char* ws = (char*)d_ws;
    size_t off = 0;
    auto alloc = [&](size_t bytes) -> void* {
        void* p = ws + off;
        off = (off + bytes + 255) & ~(size_t)255;
        return p;
    };
    _Float16* h_x0p = (_Float16*)alloc(nX0p * 2);
    _Float16* h_w1r = (_Float16*)alloc(nW1r * 2);
    _Float16* h_x1p = (_Float16*)alloc(nX1p * 2);
    _Float16* h_w2r = (_Float16*)alloc(nW2 * 2);
    float*    f_y2  = (float*)   alloc(nY2 * 4);
    _Float16* h_x2p = (_Float16*)alloc(nX2p * 2);
    _Float16* h_w3r = (_Float16*)alloc(nW3 * 2);
    float*    f_y3  = (float*)   alloc(nY3 * 4);
    _Float16* h_x3p = (_Float16*)alloc(nX3p * 2);
    _Float16* h_w4r = (_Float16*)alloc(nW4 * 2);
    float*    f_y4  = (float*)   alloc(nY4 * 4);
    _Float16* h_x4  = (_Float16*)alloc(nX4 * 2);
    _Float16* h_w5r = (_Float16*)alloc(nW5 * 2);
    _Float16* h_T   = (_Float16*)alloc(nT * 2);
    _Float16* h_ft  = (_Float16*)alloc(nFt * 2);
    float*    f_M   = (float*)   alloc(nM * 4);
    float*    f_ms  = (float*)   alloc(4096 * 4);
    float*    f_fc  = (float*)   alloc(8192 * 4);
    float*    f_mu  = (float*)   alloc(512 * 4);
    float*    f_iv  = (float*)   alloc(512 * 4);

    auto cgrid = [](size_t n) { return dim3((unsigned)((n + 255) / 256)); };

    // ---- layout conversions + pad-buffer zeroing ----
    reorder_input_kernel<<<cgrid(nX0p), 256, 0, stream>>>(x_in, h_x0p, nX0p);
    reorder_w1_kernel<<<cgrid(nW1r), 256, 0, stream>>>(w1, h_w1r);
    reorder_w_kernel<<<cgrid(nW2), 256, 0, stream>>>(w2, h_w2r, 64, 5, 5, nW2);
    reorder_w_kernel<<<cgrid(nW3), 256, 0, stream>>>(w3, h_w3r, 128, 5, 5, nW3);
    reorder_w_kernel<<<cgrid(nW4), 256, 0, stream>>>(w4, h_w4r, 256, 7, 7, nW4);
    reorder_w_kernel<<<cgrid(nW5), 256, 0, stream>>>(w5, h_w5r, 512, 4, 4, nW5);
    cvt_f32_f16<<<cgrid(nT), 256, 0, stream>>>(T, h_T, nT);
    fill_zero_f16<<<cgrid(nX1p), 256, 0, stream>>>(h_x1p, nX1p);
    fill_zero_f16<<<cgrid(nX2p), 256, 0, stream>>>(h_x2p, nX2p);
    fill_zero_f16<<<cgrid(nX3p), 256, 0, stream>>>(h_x3p, nX3p);

    // ---- conv1 (fused LeakyReLU -> padded NHWC f16 for conv2) ----
    conv_wmma_cl<4, 8, 8><<<dim3(1, (unsigned)(((size_t)Nb * 93 * 91) / 32)), 32, 0, stream>>>(
        h_x0p, h_w1r, nullptr, h_x1p, Nb, 376, 278, 64, 93, 91, 4, 3, 95, 95, 1, 2);

    // ---- conv2 + BN + LeakyReLU ----
    conv_wmma_cl<64, 5, 5><<<dim3(2, (unsigned)(((size_t)Nb * 31 * 31) / 32)), 32, 0, stream>>>(
        h_x1p, h_w2r, f_y2, nullptr, Nb, 95, 95, 128, 31, 31, 3, 3, 0, 0, 0, 0);
    bn_stats_cl<<<128 / 64, 256, 0, stream>>>(f_y2, f_mu, f_iv, Nb * 31 * 31, 128);
    bn_lrelu_cl<<<cgrid(nY2), 256, 0, stream>>>(f_y2, f_mu, f_iv, g2, b2, h_x2p,
                                                128, 31, 31, 35, 35, 2, 2, nY2);

    // ---- conv3 + BN + LeakyReLU ----
    conv_wmma_cl<128, 5, 5><<<dim3(4, (unsigned)(((size_t)Nb * 11 * 11) / 32)), 32, 0, stream>>>(
        h_x2p, h_w3r, f_y3, nullptr, Nb, 35, 35, 256, 11, 11, 3, 3, 0, 0, 0, 0);
    bn_stats_cl<<<256 / 64, 256, 0, stream>>>(f_y3, f_mu, f_iv, Nb * 11 * 11, 256);
    bn_lrelu_cl<<<cgrid(nY3), 256, 0, stream>>>(f_y3, f_mu, f_iv, g3, b3, h_x3p,
                                                256, 11, 11, 13, 13, 1, 1, nY3);

    // ---- conv4 + BN + LeakyReLU ----
    conv_wmma_cl<256, 7, 7><<<dim3(8, (unsigned)(((size_t)Nb * 4 * 4) / 32)), 32, 0, stream>>>(
        h_x3p, h_w4r, f_y4, nullptr, Nb, 13, 13, 512, 4, 4, 2, 2, 0, 0, 0, 0);
    bn_stats_cl<<<512 / 64, 256, 0, stream>>>(f_y4, f_mu, f_iv, Nb * 4 * 4, 512);
    bn_lrelu_cl<<<cgrid(nY4), 256, 0, stream>>>(f_y4, f_mu, f_iv, g4, b4, h_x4,
                                                512, 4, 4, 4, 4, 0, 0, nY4);

    // ---- conv5 as pure GEMM (K=8192, NHWC flat = (h,w,ci) matches w5r) ----
    // features -> d_out[0:16384), layout [64][256]
    conv_wmma_cl<8192, 1, 1><<<dim3(4, (unsigned)(Nb / 32)), 32, 0, stream>>>(
        h_x4, h_w5r, out, nullptr, Nb, 1, 1, 256, 1, 1, 1, 1, 0, 0, 0, 0);

    // ---- minibatch discrimination: M = T[4096,256] x features^T via WMMA ----
    cvt_f32_f16<<<cgrid(nFt), 256, 0, stream>>>(out, h_ft, nFt);
    conv_wmma_cl<256, 1, 1><<<dim3(64, (unsigned)(Nb / 32)), 32, 0, stream>>>(
        h_ft, h_T, f_M, nullptr, Nb, 1, 1, 4096, 1, 1, 1, 1, 0, 0, 0, 0);
    msum_kernel<<<dim3(64, 64), 64, 0, stream>>>(f_M, f_ms);

    // ---- fc1 + sigmoid head -> d_out[16384:16448) ----
    fc1_kernel<<<64, 128, 0, stream>>>(out, fc1_w, f_fc);
    head_kernel<<<1, 64, 0, stream>>>(f_ms, f_fc, fc2_w, out + 16384);
}
